// densityNetwork_687194767963
// MI455X (gfx1250) — matrix operand order repaired and checked by reading, not compiled, not run on hardware
//
#include <hip/hip_runtime.h>
#include <hip/hip_bf16.h>

typedef __attribute__((ext_vector_type(16))) _Float16 v16h;
typedef __attribute__((ext_vector_type(8)))  float    v8f;

#define NPTS 16384
#define KNBR 64
#define SKS  5
#define SUPPORT 0.0836f
#define EPSV 1e-12f

// ---------------------------------------------------------------------------
// Geometry: per (n,k) pair compute 4 corner bin ids (packed u32) and 4
// bilinear weights premultiplied by the neighbor mask.
// ---------------------------------------------------------------------------
__global__ __launch_bounds__(256) void cconv_geom_kernel(
    const float* __restrict__ pos, const int* __restrict__ nbr,
    const float* __restrict__ mask, unsigned* __restrict__ cidx,
    float4* __restrict__ cw)
{
    int e = blockIdx.x * blockDim.x + threadIdx.x;
    if (e >= NPTS * KNBR) return;
    int n = e >> 6;
    int nb = nbr[e];
    float m = mask[e];

    float inv = 1.0f / SUPPORT;   // (2 / extent)
    float x = (pos[nb * 3 + 0] - pos[n * 3 + 0]) * inv;
    float y = (pos[nb * 3 + 1] - pos[n * 3 + 1]) * inv;
    float z = (pos[nb * 3 + 2] - pos[n * 3 + 2]) * inv;

    // ball -> cylinder (volume preserving)
    float sq   = x * x + y * y + z * z;
    float nrm  = sqrtf(sq);
    float xy   = x * x + y * y;
    bool  pole = (1.25f * z * z) > xy;
    float s_pole = sqrtf(3.0f * nrm / (nrm + fabsf(z) + EPSV));
    float s_side = nrm * rsqrtf(fmaxf(xy, EPSV));
    float sc = pole ? s_pole : s_side;
    float cx = x * sc, cy = y * sc;
    if (sq < EPSV) { cx = 0.0f; cy = 0.0f; }

    // cylinder -> cube (disc -> square in xy)
    float sqxy = cx * cx + cy * cy;
    float rn   = sqrtf(sqxy);
    bool  use_x = fabsf(cy) <= fabsf(cx);
    float safe_cx = (fabsf(cx) > 0.0f) ? cx : 1.0f;
    float safe_cy = (fabsf(cy) > 0.0f) ? cy : 1.0f;
    const float c4pi = 1.2732395447351628f; // 4/pi
    float sgx = (cx > 0.0f) ? 1.0f : ((cx < 0.0f) ? -1.0f : 0.0f);
    float sgy = (cy > 0.0f) ? 1.0f : ((cy < 0.0f) ? -1.0f : 0.0f);
    float tx = sgx * rn, ty = sgy * rn;
    float ux = use_x ? tx : (ty * c4pi * atanf(cx / safe_cy));
    float uy = use_x ? (tx * c4pi * atanf(cy / safe_cx)) : ty;
    if (sqxy < EPSV) { ux = 0.0f; uy = 0.0f; }

    // align-corners bilinear into the 5x5 grid (depth dim is size 1)
    float gx = (ux + 1.0f) * 0.5f * (float)(SKS - 1);
    float gy = (uy + 1.0f) * 0.5f * (float)(SKS - 1);
    float x0f = fminf(fmaxf(floorf(gx), 0.0f), (float)(SKS - 1));
    float y0f = fminf(fmaxf(floorf(gy), 0.0f), (float)(SKS - 1));
    float fx = fminf(fmaxf(gx - x0f, 0.0f), 1.0f);
    float fy = fminf(fmaxf(gy - y0f, 0.0f), 1.0f);
    int x0 = (int)x0f, y0 = (int)y0f;
    int x1 = (x0 + 1 < SKS) ? x0 + 1 : SKS - 1;
    int y1 = (y0 + 1 < SKS) ? y0 + 1 : SKS - 1;

    unsigned packed = (unsigned)(y0 * SKS + x0)
                    | ((unsigned)(y0 * SKS + x1) << 8)
                    | ((unsigned)(y1 * SKS + x0) << 16)
                    | ((unsigned)(y1 * SKS + x1) << 24);
    float4 wv;
    wv.x = (1.0f - fy) * (1.0f - fx) * m;
    wv.y = (1.0f - fy) * fx * m;
    wv.z = fy * (1.0f - fx) * m;
    wv.w = fy * fx * m;
    cidx[e] = packed;
    cw[e]   = wv;
}

// ---------------------------------------------------------------------------
// Convert the three f32 kernels into zero-padded f16 B matrices [Kpad][16].
// Layer0: inner = 25 (Cin=1)  -> pad 32.  Layer1/2: inner = 400 -> pad 416.
// Layer2 has Cout=1: only column 0 nonzero.
// ---------------------------------------------------------------------------
__global__ __launch_bounds__(256) void cconv_prep_kernels(
    const float* __restrict__ k0, const float* __restrict__ k1,
    const float* __restrict__ k2, _Float16* __restrict__ B0,
    _Float16* __restrict__ B1, _Float16* __restrict__ B2)
{
    int t = blockIdx.x * blockDim.x + threadIdx.x;
    if (t < 32 * 16) {
        int r = t >> 4, o = t & 15;
        B0[t] = (_Float16)((r < 25) ? k0[r * 16 + o] : 0.0f);
    }
    if (t < 416 * 16) {
        int r = t >> 4, o = t & 15;
        B1[t] = (_Float16)((r < 400) ? k1[r * 16 + o] : 0.0f);
        B2[t] = (_Float16)((r < 400 && o == 0) ? k2[r] : 0.0f);
    }
}

// ---------------------------------------------------------------------------
// Fused layer: bin neighbor features into per-wave LDS tile, then contract
// against the f16 kernel matrix with v_wmma_f32_16x16x32_f16.
// Block = 64 threads (2 waves); each wave owns a tile of 16 queries.
// ---------------------------------------------------------------------------
template <int CIN, int KPAD, int COUTW>
__global__ __launch_bounds__(64) void cconv_layer_kernel(
    const float* __restrict__ fin, const int* __restrict__ nbr,
    const unsigned* __restrict__ cidx, const float4* __restrict__ cw,
    const _Float16* __restrict__ Bmat, float* __restrict__ fout)
{
    __shared__ float binned[2][16][KPAD];

    const int wave = threadIdx.x >> 5;
    const int lane = threadIdx.x & 31;
    const int tile = blockIdx.x * 2 + wave;
    const int n0   = tile * 16;

    // zero this wave's LDS tile (incl. zero-padding lanes of the inner dim)
    {
        float* base = &binned[wave][0][0];
        for (int i = lane; i < 16 * KPAD; i += 32) base[i] = 0.0f;
    }
    __syncthreads();

    // ---- Phase 1: binning (race-free ownership: no atomics needed) ----
    if (CIN == 16) {
        // lane pair (2q, 2q+1) owns query q; h selects an 8-channel slice
        int q = lane >> 1, h = lane & 1;
        int n = n0 + q;
        float* bq = &binned[wave][q][0];
        for (int k = 0; k < KNBR; ++k) {
            int e = n * KNBR + k;
            unsigned pk = cidx[e];
            float4 w4 = cw[e];
            int nb = nbr[e];
            const float4* fp4 = (const float4*)(fin + nb * 16) + h * 2;
            float4 fa = fp4[0], fb = fp4[1];
            float f[8] = { fa.x, fa.y, fa.z, fa.w, fb.x, fb.y, fb.z, fb.w };
            const float* wp = &w4.x;
#pragma unroll
            for (int j = 0; j < 4; ++j) {
                int bin = (pk >> (8 * j)) & 0xFF;
                float wj = wp[j];
                float* dst = bq + bin * 16 + h * 8;
#pragma unroll
                for (int c = 0; c < 8; ++c) dst[c] += wj * f[c];
            }
        }
    } else {
        // CIN == 1: lane q (< 16) owns query q entirely
        if (lane < 16) {
            int q = lane, n = n0 + q;
            float* bq = &binned[wave][q][0];
            for (int k = 0; k < KNBR; ++k) {
                int e = n * KNBR + k;
                unsigned pk = cidx[e];
                float4 w4 = cw[e];
                float f = fin[nbr[e]];
                const float* wp = &w4.x;
#pragma unroll
                for (int j = 0; j < 4; ++j)
                    bq[(pk >> (8 * j)) & 0xFF] += wp[j] * f;
            }
        }
    }
    __syncthreads();

    // ---- Phase 2: out[16 x 16] = binned[16 x KPAD] @ Bmat[KPAD x 16] ----
    v8f acc = {};
    const int M = lane & 15;        // query row of the A matrix
    const int hs = lane >> 4;       // lane-half selector (K sub-block)
    constexpr int NCHUNK = KPAD / 32;
#pragma unroll
    for (int t = 0; t < NCHUNK; ++t) {
        v16h a, b;
        // 16-bit A-matrix layout (ISA 7.12.2): lanes 0-15 hold K=0..7,16..23,
        // lanes 16-31 hold K=8..15,24..31 (2 halves per VGPR).
#pragma unroll
        for (int v = 0; v < 8; ++v) {
            int kbase = ((v < 4) ? 0 : 16) + 8 * hs + 2 * (v & 3);
            a[2 * v]     = (_Float16)binned[wave][M][t * 32 + kbase];
            a[2 * v + 1] = (_Float16)binned[wave][M][t * 32 + kbase + 1];
        }
        // B-matrix: lane L holds column N = L%16, K in the same order as A
#pragma unroll
        for (int s = 0; s < 16; ++s) {
            int kb = ((s < 8) ? 0 : 16) + 8 * hs + (s & 7);
            b[s] = Bmat[(t * 32 + kb) * 16 + M];
        }
        acc = __builtin_amdgcn_wmma_f32_16x16x32_f16(
            false, a, false, b, (short)0, acc, false, false);
    }

    // C/D layout: VGPR r -> M = r + 8*(lane/16), N = lane%16
#pragma unroll
    for (int r = 0; r < 8; ++r) {
        int m = r + 8 * hs;
        int col = lane & 15;
        if (COUTW == 16) {
            fout[(n0 + m) * 16 + col] = acc[r];
        } else {
            if (col == 0) fout[n0 + m] = acc[r];
        }
    }
}

// ---------------------------------------------------------------------------
extern "C" void kernel_launch(void* const* d_in, const int* in_sizes, int n_in,
                              void* d_out, int out_size, void* d_ws, size_t ws_size,
                              hipStream_t stream) {
    const float* positions = (const float*)d_in[0];   // [N,3]
    const float* features  = (const float*)d_in[1];   // [N,1]
    const int*   nbr       = (const int*)d_in[2];     // [N,K]
    const float* mask      = (const float*)d_in[3];   // [N,K]
    const float* kernel0   = (const float*)d_in[4];   // [1,5,5,1,16]
    const float* kernel1   = (const float*)d_in[5];   // [1,5,5,16,16]
    const float* kernel2   = (const float*)d_in[6];   // [1,5,5,16,1]
    float* out = (float*)d_out;

    char* ws = (char*)d_ws;
    unsigned* cidx = (unsigned*)ws;                               // 4 MB
    float4*   cw   = (float4*)(ws + (size_t)4194304);             // 16 MB
    float*    featA = (float*)(ws + (size_t)(4194304 + 16777216));     // 1 MB
    float*    featB = featA + (size_t)NPTS * 16;                  // 1 MB
    _Float16* B0 = (_Float16*)(ws + (size_t)(4194304 + 16777216 + 2097152));
    _Float16* B1 = B0 + 32 * 16;
    _Float16* B2 = B1 + 416 * 16;

    // 1) geometry: corner bins + mask-folded weights
    cconv_geom_kernel<<<(NPTS * KNBR) / 256, 256, 0, stream>>>(
        positions, nbr, mask, cidx, cw);

    // 2) f32 -> padded f16 kernel matrices
    cconv_prep_kernels<<<(416 * 16 + 255) / 256, 256, 0, stream>>>(
        kernel0, kernel1, kernel2, B0, B1, B2);

    // 3) three fused binning + WMMA layers (512 blocks x 2 waves = 1024 tiles)
    cconv_layer_kernel<1, 32, 16><<<NPTS / 32, 64, 0, stream>>>(
        features, nbr, cidx, cw, B0, featA);
    cconv_layer_kernel<16, 416, 16><<<NPTS / 32, 64, 0, stream>>>(
        featA, nbr, cidx, cw, B1, featB);
    cconv_layer_kernel<16, 416, 1><<<NPTS / 32, 64, 0, stream>>>(
        featB, nbr, cidx, cw, B2, out);
}
